// GGNNExtension_58523224375890
// MI455X (gfx1250) — matrix operand order repaired and checked by reading, not compiled, not run on hardware
//
#include <hip/hip_runtime.h>
#include <hip/hip_bf16.h>
#include <math.h>

// ---------------------------------------------------------------------------
// Problem constants (match reference)
// ---------------------------------------------------------------------------
#define BB 64
#define NN 8192
#define DD 512
#define MC 1000
#define TC 200
#define N_STEPS 5

typedef __attribute__((ext_vector_type(16))) __bf16 v16bf;
typedef __attribute__((ext_vector_type(8)))  __bf16 v8bf;
typedef __attribute__((ext_vector_type(8)))  float  v8f;

// ---------------------------------------------------------------------------
// f32 [M,K] row-major  ->  bf16 [M,Kp] row-major, zero-padded K..Kp
// ---------------------------------------------------------------------------
__global__ void cvt_pad_kernel(const float* __restrict__ in, __bf16* __restrict__ out,
                               int M, int K, int Kp) {
    size_t total = (size_t)M * (size_t)Kp;
    for (size_t i = (size_t)blockIdx.x * blockDim.x + threadIdx.x; i < total;
         i += (size_t)gridDim.x * blockDim.x) {
        int r = (int)(i / (size_t)Kp);
        int c = (int)(i % (size_t)Kp);
        out[i] = (c < K) ? (__bf16)in[(size_t)r * K + c] : (__bf16)0.f;
    }
}

// ---------------------------------------------------------------------------
// f32 [K,N] row-major  ->  bf16 B^T [Np,Kp] row-major (out[n*Kp+k] = in[k*N+n])
// zero-padded to Np x Kp. LDS 32x32 tiles, coalesced both directions.
// ---------------------------------------------------------------------------
__global__ __launch_bounds__(256)
void transpose_cvt_kernel(const float* __restrict__ in, __bf16* __restrict__ out,
                          int K, int N, int Kp, int Np) {
    __shared__ float tile[32][33];
    const int kBase = blockIdx.x * 32;
    const int nBase = blockIdx.y * 32;
    const int tx = threadIdx.x;   // 0..31
    const int ty = threadIdx.y;   // 0..7
#pragma unroll
    for (int i = ty; i < 32; i += 8) {
        int k = kBase + i, n = nBase + tx;
        tile[i][tx] = (k < K && n < N) ? in[(size_t)k * N + n] : 0.f;
    }
    __syncthreads();
#pragma unroll
    for (int i = ty; i < 32; i += 8) {
        int n = nBase + i, k = kBase + tx;
        if (n < Np && k < Kp) out[(size_t)n * Kp + k] = (__bf16)tile[tx][i];
    }
}

// ---------------------------------------------------------------------------
// GEMM:  C[M,N] = act( A[M,Kp]_bf16 @ (Bt[Np,Kp]_bf16)^T + bias[N] )
// A row-major bf16 (K padded, no guards); Bt = B^T row-major bf16 (N padded to
// the 256-wide block tile, no guards). 256 threads = 8 wave32, wave tile
// 32x64 (2x4 WMMA frags), block tile 64(M) x 256(N). Kp % 32 == 0, M % 64 == 0.
//
// Fragment layouts used here (wave32; contiguous k order, same for A and B):
//   A 16x32: row = lane&15; e -> K = (lane>>4)*16 + e
//            => two contiguous 8x bf16 (16B) runs at +kb and +kb+8
//   B 32x16: col = lane&15; e -> K = (lane>>4)*16 + e
//            => one contiguous 16x bf16 (32B) run from Bt row `col`
// ---------------------------------------------------------------------------
__device__ __forceinline__ float apply_act(float v, int act) {
    if (act == 1) v = 0.5f * v * (1.f + erff(v * 0.70710678118654752f));  // exact gelu
    return v;
}

__global__ __launch_bounds__(256)
void wmma_gemm_bf16_kernel(const __bf16* __restrict__ A, const __bf16* __restrict__ Bt,
                           const float* __restrict__ bias, float* __restrict__ C,
                           int M, int Ncols, int Kp, int act) {
    const int tid  = threadIdx.x;
    const int lane = tid & 31;
    const int wave = tid >> 5;
    const int wm   = wave & 1;         // 2 wave-rows   -> 64 M per block
    const int wn   = wave >> 1;        // 4 wave-cols   -> 256 N per block
    const int mBase = blockIdx.y * 64  + wm * 32;
    const int nBase = blockIdx.x * 256 + wn * 64;

    const int l15 = lane & 15;
    const int kbA = (lane >> 4) << 4;  // 0 or 16
    const int kbB = (lane >> 4) << 4;  // 0 or 16

    const __bf16* pa0 = A + (size_t)(mBase + l15) * (size_t)Kp + (size_t)kbA;
    const __bf16* pa1 = pa0 + (size_t)16 * (size_t)Kp;
    const __bf16* pb0 = Bt + (size_t)(nBase +  0 + l15) * (size_t)Kp + (size_t)kbB;
    const __bf16* pb1 = Bt + (size_t)(nBase + 16 + l15) * (size_t)Kp + (size_t)kbB;
    const __bf16* pb2 = Bt + (size_t)(nBase + 32 + l15) * (size_t)Kp + (size_t)kbB;
    const __bf16* pb3 = Bt + (size_t)(nBase + 48 + l15) * (size_t)Kp + (size_t)kbB;

    v8f acc[2][4];
#pragma unroll
    for (int mi = 0; mi < 2; ++mi)
#pragma unroll
        for (int ni = 0; ni < 4; ++ni) acc[mi][ni] = (v8f){};

    for (int k0 = 0; k0 < Kp; k0 += 32) {
        v8bf a0lo = *(const v8bf*)(pa0 + k0);
        v8bf a0hi = *(const v8bf*)(pa0 + k0 + 8);
        v8bf a1lo = *(const v8bf*)(pa1 + k0);
        v8bf a1hi = *(const v8bf*)(pa1 + k0 + 8);
        v16bf a0 = __builtin_shufflevector(a0lo, a0hi, 0,1,2,3,4,5,6,7,8,9,10,11,12,13,14,15);
        v16bf a1 = __builtin_shufflevector(a1lo, a1hi, 0,1,2,3,4,5,6,7,8,9,10,11,12,13,14,15);
        v16bf b0 = *(const v16bf*)(pb0 + k0);
        v16bf b1 = *(const v16bf*)(pb1 + k0);
        v16bf b2 = *(const v16bf*)(pb2 + k0);
        v16bf b3 = *(const v16bf*)(pb3 + k0);

        acc[0][0] = __builtin_amdgcn_wmma_f32_16x16x32_bf16(false, a0, false, b0, (short)0, acc[0][0], false, false);
        acc[0][1] = __builtin_amdgcn_wmma_f32_16x16x32_bf16(false, a0, false, b1, (short)0, acc[0][1], false, false);
        acc[0][2] = __builtin_amdgcn_wmma_f32_16x16x32_bf16(false, a0, false, b2, (short)0, acc[0][2], false, false);
        acc[0][3] = __builtin_amdgcn_wmma_f32_16x16x32_bf16(false, a0, false, b3, (short)0, acc[0][3], false, false);
        acc[1][0] = __builtin_amdgcn_wmma_f32_16x16x32_bf16(false, a1, false, b0, (short)0, acc[1][0], false, false);
        acc[1][1] = __builtin_amdgcn_wmma_f32_16x16x32_bf16(false, a1, false, b1, (short)0, acc[1][1], false, false);
        acc[1][2] = __builtin_amdgcn_wmma_f32_16x16x32_bf16(false, a1, false, b2, (short)0, acc[1][2], false, false);
        acc[1][3] = __builtin_amdgcn_wmma_f32_16x16x32_bf16(false, a1, false, b3, (short)0, acc[1][3], false, false);
    }

    // Epilogue. C/D frag: VGPR j -> row = (lane>>4)*8 + j ; col = lane&15
    const int rOff = (lane >> 4) << 3;
#pragma unroll
    for (int mi = 0; mi < 2; ++mi) {
#pragma unroll
        for (int ni = 0; ni < 4; ++ni) {
            int col = nBase + ni * 16 + l15;
            if (col < Ncols) {
                int row0 = mBase + mi * 16 + rOff;
                float bv = bias ? bias[col] : 0.f;
#pragma unroll
                for (int j = 0; j < 8; ++j) {
                    float v = apply_act(acc[mi][ni][j] + bv, act);
                    C[(size_t)(row0 + j) * (size_t)Ncols + (size_t)col] = v;
                }
            }
        }
    }
}

// ---------------------------------------------------------------------------
// Elementwise GRU kernels
// ---------------------------------------------------------------------------
__device__ __forceinline__ float sigmoidf(float x) { return 1.f / (1.f + expf(-x)); }

__global__ void gru_zr_kernel(const float* __restrict__ az, const float* __restrict__ hz,
                              const float* __restrict__ ar, const float* __restrict__ hr,
                              const float* __restrict__ bz, const float* __restrict__ br,
                              const float* __restrict__ h,
                              float* __restrict__ zout, float* __restrict__ rhout,
                              int total, int D) {
    for (int i = blockIdx.x * blockDim.x + threadIdx.x; i < total; i += gridDim.x * blockDim.x) {
        int c = i & (D - 1);
        float z = sigmoidf(az[i] + hz[i] + bz[c]);
        float r = sigmoidf(ar[i] + hr[i] + br[c]);
        float hv = h[i];
        zout[i]  = z;
        rhout[i] = r * hv;
    }
}

__global__ void gru_update_kernel(const float* __restrict__ z, const float* __restrict__ ah,
                                  const float* __restrict__ ru, const float* __restrict__ bh,
                                  float* __restrict__ h, int total, int D) {
    for (int i = blockIdx.x * blockDim.x + threadIdx.x; i < total; i += gridDim.x * blockDim.x) {
        int c = i & (D - 1);
        float zt = z[i];
        float ht = tanhf(ah[i] + ru[i] + bh[c]);
        h[i] = (1.f - zt) * h[i] + zt * ht;
    }
}

// ---------------------------------------------------------------------------
// Row softmax: one thread per row (rows=64, cols=200)
// ---------------------------------------------------------------------------
__global__ void softmax_rows_kernel(const float* __restrict__ in, float* __restrict__ out,
                                    int rows, int cols) {
    int r = blockIdx.x * blockDim.x + threadIdx.x;
    if (r >= rows) return;
    const float* x = in + (size_t)r * cols;
    float m = -3.4e38f;
    for (int c = 0; c < cols; ++c) m = fmaxf(m, x[c]);
    float s = 0.f;
    for (int c = 0; c < cols; ++c) s += expf(x[c] - m);
    float inv = 1.f / s;
    float* o = out + (size_t)r * cols;
    for (int c = 0; c < cols; ++c) o[c] = expf(x[c] - m) * inv;
}

// ---------------------------------------------------------------------------
// SqueezeExcitation over 2 channels + flatten
// ---------------------------------------------------------------------------
__global__ __launch_bounds__(256)
void se_fuse_kernel(const float* __restrict__ emb, const float* __restrict__ kg,
                    const float* __restrict__ w1, const float* __restrict__ b1,
                    const float* __restrict__ w2, const float* __restrict__ b2,
                    float* __restrict__ flat) {
    __shared__ float red0[256];
    __shared__ float red1[256];
    __shared__ float scale_sh[2];
    const int b = blockIdx.x;
    const int t = threadIdx.x;
    const float* e = emb + (size_t)b * DD;
    const float* k = kg  + (size_t)b * DD;
    float s0 = 0.f, s1 = 0.f;
    for (int d = t; d < DD; d += 256) { s0 += e[d]; s1 += k[d]; }
    red0[t] = s0; red1[t] = s1;
    __syncthreads();
    for (int off = 128; off > 0; off >>= 1) {
        if (t < off) { red0[t] += red0[t + off]; red1[t] += red1[t + off]; }
        __syncthreads();
    }
    if (t == 0) {
        float m0 = red0[0] / (float)DD;
        float m1 = red1[0] / (float)DD;
        float y0 = fmaxf(0.f, m0 * w1[0] + m1 * w1[2] + b1[0]);
        float y1 = fmaxf(0.f, m0 * w1[1] + m1 * w1[3] + b1[1]);
        scale_sh[0] = sigmoidf(y0 * w2[0] + y1 * w2[2] + b2[0]);
        scale_sh[1] = sigmoidf(y0 * w2[1] + y1 * w2[3] + b2[1]);
    }
    __syncthreads();
    float sc0 = scale_sh[0], sc1 = scale_sh[1];
    float* f = flat + (size_t)b * (2 * DD);
    for (int d = t; d < DD; d += 256) {
        f[d]      = e[d] * sc0;
        f[DD + d] = k[d] * sc1;
    }
}

// ---------------------------------------------------------------------------
// Host-side orchestration
// ---------------------------------------------------------------------------
static inline void launch_gemm(const __bf16* A, const __bf16* Bt, const float* bias, float* C,
                               int M, int N, int Kp, int act, hipStream_t s) {
    dim3 grid((N + 255) / 256, M / 64);
    wmma_gemm_bf16_kernel<<<grid, 256, 0, s>>>(A, Bt, bias, C, M, N, Kp, act);
}

static inline void launch_cvt(const float* in, __bf16* out, int M, int K, int Kp, hipStream_t s) {
    size_t total = (size_t)M * Kp;
    int blocks = (int)((total + 255) / 256);
    if (blocks > 65536) blocks = 65536;
    cvt_pad_kernel<<<blocks, 256, 0, s>>>(in, out, M, K, Kp);
}

static inline void launch_tcvt(const float* in, __bf16* out, int K, int N, int Kp, int Np,
                               hipStream_t s) {
    dim3 b(32, 8), g(Kp / 32, Np / 32);
    transpose_cvt_kernel<<<g, b, 0, s>>>(in, out, K, N, Kp, Np);
}

extern "C" void kernel_launch(void* const* d_in, const int* in_sizes, int n_in,
                              void* d_out, int out_size, void* d_ws, size_t ws_size,
                              hipStream_t stream) {
    (void)in_sizes; (void)n_in; (void)out_size; (void)ws_size;

    const float* embeddings = (const float*)d_in[0];
    const float* predicts   = (const float*)d_in[2];
    const float* node_emb   = (const float*)d_in[3];
    const float* Amat       = (const float*)d_in[4];
    const float* Wz = (const float*)d_in[5],  *Uz = (const float*)d_in[6],  *bz = (const float*)d_in[7];
    const float* Wr = (const float*)d_in[8],  *Ur = (const float*)d_in[9],  *br = (const float*)d_in[10];
    const float* Wh = (const float*)d_in[11], *Uh = (const float*)d_in[12], *bh = (const float*)d_in[13];
    const float* W_out = (const float*)d_in[14], *b_out = (const float*)d_in[15];
    const float* se_w1 = (const float*)d_in[16], *se_b1 = (const float*)d_in[17];
    const float* se_w2 = (const float*)d_in[18], *se_b2 = (const float*)d_in[19];
    const float* et_w1 = (const float*)d_in[20], *et_b1 = (const float*)d_in[21];
    const float* et_w2 = (const float*)d_in[22], *et_b2 = (const float*)d_in[23];
    const float* ml_w1 = (const float*)d_in[24], *ml_b1 = (const float*)d_in[25];
    const float* ml_w2 = (const float*)d_in[26], *ml_b2 = (const float*)d_in[27];
    const float* tl_w  = (const float*)d_in[28], *tl_b  = (const float*)d_in[29];

    const size_t ND = (size_t)NN * DD;

    // ---- workspace carve ----
    char* base = (char*)d_ws;
    auto carveF = [&](size_t n) { float* p = (float*)base;  base += ((n * 4 + 255) / 256) * 256; return p; };
    auto carveB = [&](size_t n) { __bf16* p = (__bf16*)base; base += ((n * 2 + 255) / 256) * 256; return p; };

    // f32 buffers
    float* h  = carveF(ND);
    float* a  = carveF(ND);
    float* t1 = carveF(ND);
    float* t2 = carveF(ND);
    float* t3 = carveF(ND);
    float* t4 = carveF(ND);
    float* p    = carveF((size_t)BB * TC);
    float* kg   = carveF((size_t)BB * DD);
    float* flat = carveF((size_t)BB * 2 * DD);
    float* th1  = carveF((size_t)BB * 2 * DD);
    float* th2  = carveF((size_t)BB * DD);

    // bf16 staging buffers (K padded to %32, N(transposed rows) padded to %256)
    __bf16* A_bf   = carveB((size_t)NN * NN);       // [NN, NN]
    __bf16* h_bf   = carveB(ND);                    // [NN, DD]
    __bf16* hT_bf  = carveB(ND);                    // [DD, NN]
    __bf16* a_bf   = carveB(ND);                    // [NN, DD]
    __bf16* rh_bf  = carveB(ND);                    // [NN, DD]
    __bf16* WzT = carveB((size_t)DD * DD);
    __bf16* WrT = carveB((size_t)DD * DD);
    __bf16* WhT = carveB((size_t)DD * DD);
    __bf16* UzT = carveB((size_t)DD * DD);
    __bf16* UrT = carveB((size_t)DD * DD);
    __bf16* UhT = carveB((size_t)DD * DD);
    __bf16* WoT = carveB((size_t)DD * DD);
    __bf16* etw1T = carveB((size_t)1024 * 1024);
    __bf16* etw2T = carveB((size_t)512 * 1024);
    __bf16* mlw1T = carveB((size_t)512 * 512);
    __bf16* mlw2T = carveB((size_t)1024 * 512);
    __bf16* tlwT  = carveB((size_t)256 * 1024);
    __bf16* p_bf     = carveB((size_t)BB * 224);    // [64, 224]
    __bf16* houtT_bf = carveB((size_t)DD * 224);    // [512, 224]
    __bf16* flat_bf  = carveB((size_t)BB * 1024);
    __bf16* th1_bf   = carveB((size_t)BB * 1024);
    __bf16* embx_bf  = carveB((size_t)BB * 512);
    __bf16* th2_bf   = carveB((size_t)BB * 512);
    __bf16* multi_bf = carveB((size_t)BB * 1024);

    float* out_embx  = (float*)d_out;                 // [B, D]
    float* out_multi = out_embx  + (size_t)BB * DD;   // [B, MC]
    float* out_preds = out_multi + (size_t)BB * MC;   // [B, TC]

    const int ew_blocks = (int)((ND + 255) / 256);

    // ---- one-time conversions (per launch; A + all weights) ----
    launch_cvt(Amat, A_bf, NN, NN, NN, stream);
    launch_tcvt(Wz, WzT, DD, DD, DD, DD, stream);
    launch_tcvt(Wr, WrT, DD, DD, DD, DD, stream);
    launch_tcvt(Wh, WhT, DD, DD, DD, DD, stream);
    launch_tcvt(Uz, UzT, DD, DD, DD, DD, stream);
    launch_tcvt(Ur, UrT, DD, DD, DD, DD, stream);
    launch_tcvt(Uh, UhT, DD, DD, DD, DD, stream);
    launch_tcvt(W_out, WoT, DD, DD, DD, DD, stream);
    launch_tcvt(et_w1, etw1T, 1024, 1024, 1024, 1024, stream);
    launch_tcvt(et_w2, etw2T, 1024, 512, 1024, 512, stream);
    launch_tcvt(ml_w1, mlw1T, 512, 512, 512, 512, stream);
    launch_tcvt(ml_w2, mlw2T, 512, 1000, 512, 1024, stream);
    launch_tcvt(tl_w,  tlwT,  1000, 200, 1024, 256, stream);

    // h = node_emb
    hipMemcpyAsync(h, node_emb, ND * sizeof(float), hipMemcpyDeviceToDevice, stream);

    // ---- GGNN: 5 propagation + GRU steps ----
    for (int step = 0; step < N_STEPS; ++step) {
        launch_cvt(h, h_bf, NN, DD, DD, stream);
        launch_tcvt(h, hT_bf, NN, DD, NN, DD, stream);            // h^T [DD, NN]
        launch_gemm(A_bf, hT_bf, nullptr, a, NN, DD, NN, 0, stream);  // a = A @ h
        launch_cvt(a, a_bf, NN, DD, DD, stream);
        launch_gemm(a_bf, WzT, nullptr, t1, NN, DD, DD, 0, stream);   // az
        launch_gemm(h_bf, UzT, nullptr, t2, NN, DD, DD, 0, stream);   // hz
        launch_gemm(a_bf, WrT, nullptr, t3, NN, DD, DD, 0, stream);   // ar
        launch_gemm(h_bf, UrT, nullptr, t4, NN, DD, DD, 0, stream);   // hr
        gru_zr_kernel<<<ew_blocks, 256, 0, stream>>>(t1, t2, t3, t4, bz, br, h,
                                                     t1 /*z*/, t3 /*r*h*/, (int)ND, DD);
        launch_gemm(a_bf, WhT, nullptr, t2, NN, DD, DD, 0, stream);   // ah
        launch_cvt(t3, rh_bf, NN, DD, DD, stream);
        launch_gemm(rh_bf, UhT, nullptr, t4, NN, DD, DD, 0, stream);  // (r*h) @ Uh
        gru_update_kernel<<<ew_blocks, 256, 0, stream>>>(t1, t2, t4, bh, h, (int)ND, DD);
    }

    // h_out = h @ W_out + b_out  (stored in `a`)
    launch_cvt(h, h_bf, NN, DD, DD, stream);
    launch_gemm(h_bf, WoT, b_out, a, NN, DD, DD, 0, stream);

    // p = softmax(predicts); kg = p @ h_out[:TC]
    softmax_rows_kernel<<<1, 64, 0, stream>>>(predicts, p, BB, TC);
    launch_cvt(p, p_bf, BB, TC, 224, stream);
    launch_tcvt(a, houtT_bf, TC, DD, 224, DD, stream);    // (h_out[:200])^T -> [512, 224]
    launch_gemm(p_bf, houtT_bf, nullptr, kg, BB, DD, 224, 0, stream);

    // SqueezeExcitation + flatten
    se_fuse_kernel<<<BB, 256, 0, stream>>>(embeddings, kg, se_w1, se_b1, se_w2, se_b2, flat);

    // SimpleHead
    launch_cvt(flat, flat_bf, BB, 1024, 1024, stream);
    launch_gemm(flat_bf, etw1T, et_b1, th1, BB, 1024, 1024, 1, stream);     // gelu
    launch_cvt(th1, th1_bf, BB, 1024, 1024, stream);
    launch_gemm(th1_bf, etw2T, et_b2, out_embx, BB, DD, 1024, 0, stream);   // embedding_x
    launch_cvt(out_embx, embx_bf, BB, DD, DD, stream);
    launch_gemm(embx_bf, mlw1T, ml_b1, th2, BB, DD, DD, 1, stream);         // gelu
    launch_cvt(th2, th2_bf, BB, DD, DD, stream);
    launch_gemm(th2_bf, mlw2T, ml_b2, out_multi, BB, MC, DD, 0, stream);    // multi
    launch_cvt(out_multi, multi_bf, BB, MC, 1024, stream);
    launch_gemm(multi_bf, tlwT, tl_b, out_preds, BB, TC, 1024, 0, stream);  // preds
}
